// FFTConvModule_26774826123741
// MI455X (gfx1250) — matrix operand order repaired and checked by reading, not compiled, not run on hardware
//
#include <hip/hip_runtime.h>

typedef __attribute__((ext_vector_type(16))) __bf16 bf16x16;
typedef __attribute__((ext_vector_type(8)))  float  f32x8;
typedef __attribute__((ext_vector_type(4)))  float  f32x4;
typedef __attribute__((ext_vector_type(8)))  unsigned int u32x8;

namespace {
constexpr int kB = 16;
constexpr int kD = 1024;
constexpr int kL = 4096;
constexpr int kThreads = 256;      // 8 waves (wave32)
constexpr int kXPitch = 2049;      // bf16-pair dwords per batch row: 2048 + 1 pad (bank spread)
constexpr int kFrevN = 4096 + 32;  // reversed filter + zero tail
}

__device__ __forceinline__ unsigned bf16_rne(float v) {
  unsigned u = __builtin_bit_cast(unsigned, v);
  return (u + 0x7FFFu + ((u >> 16) & 1u)) >> 16;  // round-to-nearest-even f32->bf16
}

// y[b,d,l] = sum_{t<=l} f[d,t] * x[b,d,l-t]
// Per d:  Y (4096x16) = T(f_d) (lower-tri Toeplitz) @ X (4096x16), tiled as
// 16x16x32 bf16 WMMAs.  Toeplitz block at (row 16m, col 32j) depends only on
// dd = m - 2j, so tiles m, m+2, m+4, m+6 at columns j..j+3 share one A fragment:
// 4 WMMAs per A load, B fragments in a rolling ring of 4.
__global__ __launch_bounds__(kThreads, 1)
void fftconv_toeplitz_wmma(const float* __restrict__ x,
                           const float* __restrict__ filt,
                           float* __restrict__ y) {
  // frev declared FIRST so A-fragment addresses stay within the 16-bit DS
  // immediate-offset range (no per-load v_add in the hot loop).
  __shared__ unsigned short frev[kFrevN];   // frev[i] = bf16(f[d, 4095 - i]); tail = 0
  __shared__ unsigned xp[kB * kXPitch];     // x[b, d, :] as packed bf16 pairs

  const int d = blockIdx.x;
  const int tid = threadIdx.x;

  // Stage reversed filter (bf16) + zero tail (read-once -> non-temporal).
  for (int j = tid; j < kFrevN; j += kThreads) {
    float v = (j < kL) ? __builtin_nontemporal_load(&filt[(size_t)d * kL + (kL - 1 - j)])
                       : 0.0f;
    frev[j] = (unsigned short)bf16_rne(v);
  }
  // Stage x slab for this d: 16 batches x 4096 -> packed bf16 pairs (b128 NT reads).
  for (int idx = tid; idx < kB * (kL / 4); idx += kThreads) {
    const int b = idx >> 10;        // / 1024
    const int q = idx & 1023;
    const f32x4* px = (const f32x4*)(x + (((size_t)b * kD + d) * kL + 4 * q));
    const f32x4 v = __builtin_nontemporal_load(px);
    xp[b * kXPitch + 2 * q]     = bf16_rne(v.x) | (bf16_rne(v.y) << 16);
    xp[b * kXPitch + 2 * q + 1] = bf16_rne(v.z) | (bf16_rne(v.w) << 16);
  }
  __syncthreads();

  const int lane = tid & 31;
  const int wv   = tid >> 5;
  const int nrow = lane & 15;   // A: row M;  B/C/D: column N (= batch b)
  const int khi  = lane >> 4;   // which K half this lane holds

  // Per-VGPR base K for the 16x32 bf16 A fragment (contiguous K order, same as B):
  // v0..7: K 0..15 (lanes<16) / 16..31 (lanes>=16).
  int akk[8];
  #pragma unroll
  for (int v = 0; v < 8; ++v) akk[v] = khi * 16 + v * 2;

  const unsigned short* pa0 = &frev[(kL - 1) - nrow];          // A(dd) lives at pa0 - 16*dd
  const unsigned*       pb  = &xp[nrow * kXPitch + 8 * khi];   // B(j)  lives at pb + 16*j

  // A(dd)[r,k] = f[16*dd + r - k] = pa[k]  (16 consecutive bf16 per lane).
  auto loadA_at = [&](const unsigned short* pa) -> bf16x16 {
    u32x8 au;
    #pragma unroll
    for (int v = 0; v < 8; ++v)
      au[v] = (unsigned)pa[akk[v]] | ((unsigned)pa[akk[v] + 1] << 16);
    return __builtin_bit_cast(bf16x16, au);
  };
  // Clamped variant for the padded epilogue: dd < 0 -> read the zero tail.
  auto loadA_dd = [&](int dd) -> bf16x16 {
    const int G = (dd >= 0) ? ((kL - 1) - 16 * dd - nrow) : kL;
    u32x8 au;
    #pragma unroll
    for (int v = 0; v < 8; ++v) {
      const int ia = G + akk[v];
      au[v] = (unsigned)frev[ia] | ((unsigned)frev[ia + 1] << 16);
    }
    return __builtin_bit_cast(bf16x16, au);
  };
  auto loadB = [&](int j) -> u32x8 {
    u32x8 bu;
    #pragma unroll
    for (int v = 0; v < 8; ++v) bu[v] = pb[16 * j + v];
    return bu;
  };
  auto mm = [&](const bf16x16& a, const u32x8& bu, f32x8 c) -> f32x8 {
    return __builtin_amdgcn_wmma_f32_16x16x32_bf16(
        false, a, false, __builtin_bit_cast(bf16x16, bu), (short)0, c, false, false);
  };

  // 64 groups of 4 same-parity tiles {base, base+2, base+4, base+6};
  // base = 8*(gi>>1) + (gi&1).  Mirrored assignment balances triangular work.
  for (int g = 0; g < 8; ++g) {
    const int gi = (g < 4) ? (wv + 8 * g) : (63 - wv - 8 * (g - 4));
    const int base = __builtin_amdgcn_readfirstlane(8 * (gi >> 1) + (gi & 1));
    const unsigned short* paB = pa0 - 16 * base;   // A(base - 2j) = paB + 32*j

    f32x8 c0 = {}, c1 = {}, c2 = {}, c3 = {};
    u32x8 B0 = loadB(0), B1 = loadB(1), B2 = loadB(2), B3;
    bf16x16 A;

    // Prologue: tile base+2t needs extra columns j' = 0..t-1  (A(base+2t-2j')).
    A = loadA_at(paB - 32);  c1 = mm(A, B0, c1);  c2 = mm(A, B1, c2);  c3 = mm(A, B2, c3);
    A = loadA_at(paB - 64);  c2 = mm(A, B0, c2);  c3 = mm(A, B1, c3);
    A = loadA_at(paB - 96);  c3 = mm(A, B0, c3);

    // Main loop: iteration j uses A(base-2j) with B(j..j+3).
    const int jend  = base >> 1;              // inclusive; dd reaches 0 or 1
    const int jfull = (jend + 1) & ~3;        // clamp-free full blocks
    for (int j = 0; j < jfull; j += 4) {
      B3 = loadB(j + 3);
      A = loadA_at(paB + 32 * j);
      c0 = mm(A, B0, c0); c1 = mm(A, B1, c1); c2 = mm(A, B2, c2); c3 = mm(A, B3, c3);
      B0 = loadB(j + 4);
      A = loadA_at(paB + 32 * (j + 1));
      c0 = mm(A, B1, c0); c1 = mm(A, B2, c1); c2 = mm(A, B3, c2); c3 = mm(A, B0, c3);
      B1 = loadB(j + 5);
      A = loadA_at(paB + 32 * (j + 2));
      c0 = mm(A, B2, c0); c1 = mm(A, B3, c1); c2 = mm(A, B0, c2); c3 = mm(A, B1, c3);
      B2 = loadB(j + 6);
      A = loadA_at(paB + 32 * (j + 3));
      c0 = mm(A, B3, c0); c1 = mm(A, B0, c1); c2 = mm(A, B1, c2); c3 = mm(A, B2, c3);
    }
    // Epilogue: one clamped block covering the 1..3 remaining columns
    // (dd < 0 sub-steps read the zero tail and contribute nothing).
    if (jfull <= jend) {
      const int j = jfull;
      B3 = loadB(j + 3);
      A = loadA_dd(base - 2 * j);
      c0 = mm(A, B0, c0); c1 = mm(A, B1, c1); c2 = mm(A, B2, c2); c3 = mm(A, B3, c3);
      B0 = loadB(j + 4);
      A = loadA_dd(base - 2 * (j + 1));
      c0 = mm(A, B1, c0); c1 = mm(A, B2, c1); c2 = mm(A, B3, c2); c3 = mm(A, B0, c3);
      B1 = loadB(j + 5);
      A = loadA_dd(base - 2 * (j + 2));
      c0 = mm(A, B2, c0); c1 = mm(A, B3, c1); c2 = mm(A, B0, c2); c3 = mm(A, B1, c3);
      B2 = loadB(j + 6);
      A = loadA_dd(base - 2 * (j + 3));
      c0 = mm(A, B3, c0); c1 = mm(A, B0, c1); c2 = mm(A, B1, c2); c3 = mm(A, B2, c3);
    }

    // D layout: N = nrow (=b), M = r + 8*khi -> 8 consecutive f32 (32B aligned) per lane.
    const size_t ybase = ((size_t)nrow * kD + d) * kL + 8 * khi;
    const f32x8 cc[4] = {c0, c1, c2, c3};
    #pragma unroll
    for (int t = 0; t < 4; ++t) {
      float* py = y + ybase + 16 * (base + 2 * t);
      const f32x4 lo = {cc[t][0], cc[t][1], cc[t][2], cc[t][3]};
      const f32x4 hi = {cc[t][4], cc[t][5], cc[t][6], cc[t][7]};
      __builtin_nontemporal_store(lo, (f32x4*)py);
      __builtin_nontemporal_store(hi, (f32x4*)(py + 4));
    }
  }
}

extern "C" void kernel_launch(void* const* d_in, const int* in_sizes, int n_in,
                              void* d_out, int out_size, void* d_ws, size_t ws_size,
                              hipStream_t stream) {
  (void)in_sizes; (void)n_in; (void)out_size; (void)d_ws; (void)ws_size;
  const float* x = (const float*)d_in[0];     // [16, 1024, 4096] f32
  const float* f = (const float*)d_in[1];     // [1024, 4096] f32
  float* y = (float*)d_out;                   // [16, 1024, 4096] f32
  fftconv_toeplitz_wmma<<<dim3(kD), dim3(kThreads), 0, stream>>>(x, f, y);
}
